// GridNetBlock_33689723470009
// MI455X (gfx1250) — matrix operand order repaired and checked by reading, not compiled, not run on hardware
//
#include <hip/hip_runtime.h>
#include <hip/hip_bf16.h>
#include <math.h>

typedef __attribute__((ext_vector_type(16))) __bf16 v16bf;
typedef __attribute__((ext_vector_type(8)))  __bf16 v8bf;
typedef __attribute__((ext_vector_type(8)))  float  v8f;

union V16 { v16bf v; v8bf h[2]; };

#define DEV __device__ __forceinline__

// ---------------- problem sizes ----------------
constexpr int B_ = 2, C_ = 48, T_ = 600, F_ = 65, H_ = 192, G_ = 768;
constexpr int KS_ = 4, L1_ = 62, L2_ = 597, BT_ = 1200, BF_ = 130;
constexpr int NH_ = 4, E_ = 8, VD_ = 12, NHB_ = 8;
constexpr int DQP_ = 544, DV_ = 780, DVP_ = 784, TP_ = 640, LA_ = 100;
constexpr float EPS_ = 1e-5f;

DEV float sigm(float x){ return 1.0f/(1.0f+__expf(-x)); }

DEV v8f wmma_bf16(v16bf a, v16bf b, v8f c){
  return __builtin_amdgcn_wmma_f32_16x16x32_bf16(false, a, false, b, (short)0, c,
                                                 false, false);
}

DEV float block_sum(float v, float* red){
  int tid = threadIdx.x;
  red[tid] = v; __syncthreads();
  for (int s = 128; s > 0; s >>= 1){
    if (tid < s) red[tid] += red[tid+s];
    __syncthreads();
  }
  float r = red[0]; __syncthreads();
  return r;
}

// ---- K0: build fragment-swizzled Wc = [wih^T ; whh^T] (bf16) and bc -------
// Swizzled layout: Wc[(kt*48+nt)*512 + lane*16 + e] = logical[k][n],
//   k = kt*32 + (e&7) + (e>=8 ? 16 : 0) + (lane>=16 ? 8 : 0), n = nt*16 + (lane&15)
__global__ void prep_weights(const float* __restrict__ wih_f, const float* __restrict__ whh_f,
                             const float* __restrict__ bih_f, const float* __restrict__ bhh_f,
                             const float* __restrict__ wih_b, const float* __restrict__ whh_b,
                             const float* __restrict__ bih_b, const float* __restrict__ bhh_b,
                             const float* __restrict__ wih_i, const float* __restrict__ whh_i,
                             const float* __restrict__ bih_i, const float* __restrict__ bhh_i,
                             __bf16* __restrict__ WcF, __bf16* __restrict__ WcB,
                             __bf16* __restrict__ WcI,
                             float* __restrict__ bcF, float* __restrict__ bcB,
                             float* __restrict__ bcI)
{
  const int WSZ = 384*G_;
  size_t idx = (size_t)blockIdx.x*256 + threadIdx.x;
  if (idx < (size_t)3*WSZ){
    int which = (int)(idx / WSZ);
    int r = (int)(idx % WSZ);
    int e    = r & 15;
    int lane = (r >> 4) & 31;
    int nt   = (r >> 9) % 48;
    int kt   = r / 24576;
    int k = kt*32 + (e & 7) + ((e >> 3) << 4) + ((lane >= 16) ? 8 : 0);
    int n = nt*16 + (lane & 15);
    const float* wih = which==0 ? wih_f : (which==1 ? wih_b : wih_i);
    const float* whh = which==0 ? whh_f : (which==1 ? whh_b : whh_i);
    float v = (k < 192) ? wih[(size_t)n*192 + k] : whh[(size_t)n*192 + (k-192)];
    __bf16* Wc = which==0 ? WcF : (which==1 ? WcB : WcI);
    Wc[r] = (__bf16)v;
  } else if (idx < (size_t)3*WSZ + 3*G_){
    int r = (int)(idx - (size_t)3*WSZ);
    int which = r / G_, n = r % G_;
    const float* bi = which==0 ? bih_f : (which==1 ? bih_b : bih_i);
    const float* bh = which==0 ? bhh_f : (which==1 ? bhh_b : bhh_i);
    float* bc = which==0 ? bcF : (which==1 ? bcB : bcI);
    bc[n] = bi[n] + bh[n];
  }
}

// ---- K1: LN over C + unfold (intra), one block per (b,t) ------------------
__global__ void ln_unfold_intra(const float* __restrict__ x, const float* __restrict__ g,
                                const float* __restrict__ bv, __bf16* __restrict__ u)
{
  __shared__ float zs[C_*F_];
  __shared__ float mu[F_], ri[F_];
  int blk = blockIdx.x; int b = blk / T_, t = blk % T_;
  int tid = threadIdx.x;
  for (int idx = tid; idx < C_*F_; idx += 256){
    int c = idx / F_, f = idx % F_;
    zs[idx] = x[(((size_t)b*C_ + c)*T_ + t)*F_ + f];
  }
  __syncthreads();
  if (tid < F_){
    float s = 0.f;
    for (int c = 0; c < C_; ++c) s += zs[c*F_ + tid];
    float m = s / C_;
    float v = 0.f;
    for (int c = 0; c < C_; ++c){ float d = zs[c*F_ + tid] - m; v += d*d; }
    mu[tid] = m; ri[tid] = rsqrtf(v / C_ + EPS_);
  }
  __syncthreads();
  for (int idx = tid; idx < C_*F_; idx += 256){
    int c = idx / F_, f = idx % F_;
    zs[idx] = (zs[idx] - mu[f]) * ri[f] * g[c] + bv[c];
  }
  __syncthreads();
  for (int idx = tid; idx < L1_*192; idx += 256){
    int l = idx / 192, k = idx % 192;
    int c = k >> 2, kk = k & 3;
    u[((size_t)blk*L1_ + l)*192 + k] = (__bf16)zs[c*F_ + l + kk];
  }
}

// ---- K2: fused WMMA LSTM --------------------------------------------------
// per step: gates(16x768) = [u_l ; h](16x384) @ Wc(384x768) + bc
__global__ void lstm_kernel(const __bf16* __restrict__ u,   // (N, L, 192)
                            const __bf16* __restrict__ Wc,  // swizzled (12,48,32,16)
                            const float*  __restrict__ bc,  // (768)
                            const float*  __restrict__ h0,  // (N,192) or null
                            const float*  __restrict__ c0,  // (N,192) or null
                            float* __restrict__ y,          // (N, L, ldOut)
                            int N, int L, int ldOut, int outOff, int backward)
{
  __shared__ __align__(16) __bf16 hbf[16*192];
  __shared__ float  zbuf[16*G_];
  int row0 = blockIdx.x * 16;
  int tid = threadIdx.x, lane = tid & 31, wave = tid >> 5;

  float creg[12];
  #pragma unroll
  for (int ii = 0; ii < 12; ++ii){
    int idx = tid + ii*256;
    int m = idx / 192, j = idx % 192;
    int r = row0 + m;
    float hv = 0.f, cv = 0.f;
    if (h0 && r < N){ hv = h0[(size_t)r*192 + j]; cv = c0[(size_t)r*192 + j]; }
    creg[ii] = cv;
    hbf[idx] = (__bf16)hv;
  }
  __syncthreads();

  int m = lane & 15;
  int koff = (lane >= 16) ? 8 : 0;
  int rr = row0 + m; if (rr >= N) rr = 0;

  // per-lane bias values for this wave's 6 n-tiles (loop-invariant, tiny)
  float bcv[6];
  #pragma unroll
  for (int q = 0; q < 6; ++q) bcv[q] = bc[(wave*6 + q)*16 + (lane & 15)];

  // single base address for ALL weight loads of this lane; per-fragment
  // displacements become compile-time instruction offsets (fit in IOFFSET).
  const __bf16* wlane = Wc + ((size_t)wave*6)*512 + (size_t)lane*16;

  for (int s = 0; s < L; ++s){
    int lpos = backward ? (L-1-s) : s;
    const __bf16* up = u + ((size_t)rr*L + lpos)*192;
    if (s + 1 < L){
      int lnext = backward ? (L-2-s) : (s+1);
      __builtin_prefetch(u + ((size_t)rr*L + lnext)*192, 0, 0);
    }
    // Launder an integer offset (actually 0) each step: weight loads cannot
    // be hoisted out of the loop (would spill), but pointer provenance stays
    // rooted at the Wc kernarg so addrspace inference keeps GLOBAL loads
    // (flat_load would false-couple the weight stream to DScnt).
    size_t wz = 0;
    asm volatile("" : "+s"(wz));
    const __bf16* w0 = wlane + wz;
    v8f acc[6];
    #pragma unroll
    for (int q = 0; q < 6; ++q) acc[q] = (v8f){};
    #pragma unroll
    for (int kt = 0; kt < 12; ++kt){
      V16 a;
      int kbase = kt*32 + koff;
      if (kt < 6){                               // input part: global, wide
        a.h[0] = *(const v8bf*)(up + kbase);
        a.h[1] = *(const v8bf*)(up + kbase + 16);
      } else {                                   // hidden part: LDS, wide
        const __bf16* hp = hbf + m*192 + (kbase - 192);
        a.h[0] = *(const v8bf*)(hp);
        a.h[1] = *(const v8bf*)(hp + 16);
      }
      #pragma unroll
      for (int q = 0; q < 6; ++q){
        const __bf16* wp2 = w0 + kt*24576 + q*512;   // constant displacement
        V16 bb;
        bb.h[0] = *(const v8bf*)(wp2);
        bb.h[1] = *(const v8bf*)(wp2 + 8);
        acc[q] = wmma_bf16(a.v, bb.v, acc[q]);
      }
    }
    #pragma unroll
    for (int q = 0; q < 6; ++q){
      int nt = wave*6 + q;
      #pragma unroll
      for (int i = 0; i < 8; ++i){
        int mm = i + ((lane >= 16) ? 8 : 0);
        int n = nt*16 + (lane & 15);
        zbuf[mm*G_ + n] = acc[q][i] + bcv[q];
      }
    }
    __syncthreads();
    #pragma unroll
    for (int ii = 0; ii < 12; ++ii){
      int idx = tid + ii*256;
      int mm = idx / 192, j = idx % 192;
      float zi = zbuf[mm*G_ + j];
      float zf = zbuf[mm*G_ + 192 + j];
      float zg = zbuf[mm*G_ + 384 + j];
      float zo = zbuf[mm*G_ + 576 + j];
      float cv = creg[ii];
      cv = sigm(zf)*cv + sigm(zi)*tanhf(zg);
      float hv = sigm(zo)*tanhf(cv);
      creg[ii] = cv;
      hbf[idx] = (__bf16)hv;
      int r = row0 + mm;
      if (r < N) y[((size_t)r*L + lpos)*ldOut + outOff + j] = hv;
    }
    __syncthreads();
  }
}

// ---- K3: transposed conv + residual ---------------------------------------
__global__ void convT_res(const float* __restrict__ yin, const float* __restrict__ wT,
                          const float* __restrict__ bias, const float* __restrict__ res,
                          float* __restrict__ outp, int Ll, int J, int mode)
{
  size_t idx = (size_t)blockIdx.x*256 + threadIdx.x;
  if (idx >= (size_t)B_*C_*T_*F_) return;
  int f = (int)(idx % F_); size_t r0 = idx / F_;
  int t = (int)(r0 % T_); r0 /= T_;
  int c = (int)(r0 % C_); int b = (int)(r0 / C_);
  int n, q;
  if (mode == 0){ n = b*T_ + t; q = f; } else { n = b*F_ + f; q = t; }
  float acc = bias[c];
  for (int k = 0; k < KS_; ++k){
    int l = q - k;
    if (l < 0 || l >= Ll) continue;
    const float* yrow = yin + ((size_t)n*Ll + l)*J;
    for (int j = 0; j < J; ++j)
      acc += yrow[j] * wT[((size_t)j*C_ + c)*KS_ + k];
  }
  outp[idx] = res[idx] + acc;
}

// ---- K4: LN over C (inter), chunks of 150 t per block ---------------------
__global__ void ln_inter(const float* __restrict__ x1, const float* __restrict__ g,
                         const float* __restrict__ bv, __bf16* __restrict__ zn)
{
  __shared__ float xs[C_*150];
  __shared__ float mu[150], ri[150];
  int blk = blockIdx.x;
  int chunk = blk & 3; int bf = blk >> 2;
  int b = bf / F_, f = bf % F_;
  int t0 = chunk*150;
  int tid = threadIdx.x;
  for (int idx = tid; idx < C_*150; idx += 256){
    int c = idx / 150, tt = idx % 150;
    xs[idx] = x1[(((size_t)b*C_ + c)*T_ + t0 + tt)*F_ + f];
  }
  __syncthreads();
  for (int tt = tid; tt < 150; tt += 256){
    float s = 0.f;
    for (int c = 0; c < C_; ++c) s += xs[c*150 + tt];
    float mm = s / C_;
    float v = 0.f;
    for (int c = 0; c < C_; ++c){ float d = xs[c*150 + tt] - mm; v += d*d; }
    mu[tt] = mm; ri[tt] = rsqrtf(v / C_ + EPS_);
  }
  __syncthreads();
  for (int idx = tid; idx < C_*150; idx += 256){
    int c = idx / 150, tt = idx % 150;
    zn[((size_t)bf*C_ + c)*T_ + t0 + tt] =
        (__bf16)((xs[idx] - mu[tt])*ri[tt]*g[c] + bv[c]);
  }
}

// ---- K5: unfold (inter) ---------------------------------------------------
__global__ void unfold_inter(const __bf16* __restrict__ zn, __bf16* __restrict__ u2)
{
  size_t idx = (size_t)blockIdx.x*256 + threadIdx.x;
  if (idx >= (size_t)BF_*L2_*192) return;
  int k = (int)(idx % 192); size_t r0 = idx / 192;
  int l = (int)(r0 % L2_); int bf = (int)(r0 / L2_);
  int c = k >> 2, kk = k & 3;
  u2[idx] = zn[((size_t)bf*C_ + c)*T_ + l + kk];
}

// ---- K6: Q/K/V head projection + PReLU + LN -> padded bf16 ----------------
__global__ void proj_head_ln(const float* __restrict__ outp, const float* __restrict__ w,
                             const float* __restrict__ bias, const float* __restrict__ g,
                             const float* __restrict__ bet, const float* __restrict__ aptr,
                             __bf16* __restrict__ dst, int O, int Dpad,
                             int transposed, int ldT)
{
  __shared__ float buf[VD_*F_];  // max 780
  __shared__ float red[256];
  int blk = blockIdx.x;
  int t = blk % T_; int hb2 = blk / T_;
  int b = hb2 % B_; int h = hb2 / B_;
  int tid = threadIdx.x;
  float a = aptr[0];
  int D = O*F_;
  for (int idx = tid; idx < D; idx += 256){
    int o = idx / F_, f = idx % F_;
    float s = bias[h*O + o];
    const float* wr = w + ((size_t)h*O + o)*C_;
    for (int c = 0; c < C_; ++c)
      s += wr[c] * outp[(((size_t)b*C_ + c)*T_ + t)*F_ + f];
    buf[idx] = (s >= 0.f) ? s : a*s;
  }
  __syncthreads();
  float ls = 0.f;
  for (int idx = tid; idx < D; idx += 256) ls += buf[idx];
  float mean = block_sum(ls, red) / D;
  float lv = 0.f;
  for (int idx = tid; idx < D; idx += 256){ float d = buf[idx] - mean; lv += d*d; }
  float ri = rsqrtf(block_sum(lv, red) / D + EPS_);
  int hb = h*B_ + b;
  for (int idx = tid; idx < Dpad; idx += 256){
    float v = 0.f;
    if (idx < D){
      int o = idx / F_, f = idx % F_;
      v = (buf[idx] - mean)*ri * g[((size_t)h*O + o)*F_ + f]
          + bet[((size_t)h*O + o)*F_ + f];
    }
    if (!transposed) dst[((size_t)hb*T_ + t)*Dpad + idx] = (__bf16)v;
    else             dst[((size_t)hb*Dpad + idx)*ldT + t] = (__bf16)v;
  }
}

// ---- K7: banded causal attention (WMMA flash style) -----------------------
__global__ void attn_kernel(const __bf16* __restrict__ Qb, const __bf16* __restrict__ Kb,
                            const __bf16* __restrict__ Vt, float* __restrict__ VoR)
{
  __shared__ float  S[16*128];
  __shared__ __align__(16) __bf16 P[16*128];
  int it = blockIdx.x, hb = blockIdx.y;
  int i0 = it*16;
  int jm = i0 - (LA_-1);
  int jmin = (jm > 0) ? (jm & ~7) : 0;   // 8-aligned band start, width <= 122
  int tid = threadIdx.x, lane = tid & 31, wave = tid >> 5;
  int b = hb % B_, h = hb / B_;
  int m = lane & 15;
  int koffA = (lane >= 16) ? 8 : 0;
  int koffB = (lane >= 16) ? 16 : 0;

  // S = Q * K^T over the band
  {
    int nt = wave;
    int irow = i0 + m; if (irow > T_-1) irow = T_-1;
    int j = jmin + nt*16 + m; if (j > T_-1) j = T_-1;
    const __bf16* qrow = Qb + ((size_t)hb*T_ + irow)*DQP_;
    const __bf16* krow = Kb + ((size_t)hb*T_ + j)*DQP_;
    v8f acc = (v8f){};
    #pragma unroll
    for (int kt = 0; kt < DQP_/32; ++kt){
      V16 a, bb;
      int kbase = kt*32 + koffA;
      a.h[0] = *(const v8bf*)(qrow + kbase);
      a.h[1] = *(const v8bf*)(qrow + kbase + 16);
      const __bf16* kp = krow + kt*32 + koffA;
      bb.h[0] = *(const v8bf*)(kp);
      bb.h[1] = *(const v8bf*)(kp + 16);
      acc = wmma_bf16(a.v, bb.v, acc);
    }
    const float scale = 0.04385290096535147f;  // 1/sqrt(520)
    #pragma unroll
    for (int i = 0; i < 8; ++i){
      int mm = i + ((lane >= 16) ? 8 : 0);
      int jj = nt*16 + (lane & 15);
      int gi = i0 + mm, gj = jmin + jj;
      bool ok = (gi < T_) && (gj < T_) && (gj <= gi) && (gj > gi - LA_);
      S[mm*128 + jj] = ok ? acc[i]*scale : -1e30f;
    }
  }
  __syncthreads();
  // row softmax (one thread per row)
  if (tid < 16){
    float mx = -1e30f;
    for (int j2 = 0; j2 < 128; ++j2) mx = fmaxf(mx, S[tid*128 + j2]);
    float sum = 0.f;
    for (int j2 = 0; j2 < 128; ++j2){
      float e_ = __expf(S[tid*128 + j2] - mx);
      S[tid*128 + j2] = e_; sum += e_;
    }
    float inv = 1.0f / sum;
    for (int j2 = 0; j2 < 128; ++j2) P[tid*128 + j2] = (__bf16)(S[tid*128 + j2]*inv);
  }
  __syncthreads();
  // O = P(16x128) @ V(128x780), V stored j-major (hb, d, j) with ldT=TP_
  for (int r = 0; r < 7; ++r){
    int nt = wave + r*8;
    if (nt*16 >= DVP_) break;
    int n = nt*16 + (lane & 15);
    const __bf16* vrow = Vt + ((size_t)hb*DVP_ + n)*TP_ + jmin;
    v8f acc = (v8f){};
    #pragma unroll
    for (int kt = 0; kt < 4; ++kt){
      V16 a, bb;
      int kbase = kt*32 + koffA;
      a.h[0] = *(const v8bf*)(P + m*128 + kbase);
      a.h[1] = *(const v8bf*)(P + m*128 + kbase + 16);
      const __bf16* vp = vrow + kt*32 + koffA;   // masked cols have P==0
      bb.h[0] = *(const v8bf*)(vp);
      bb.h[1] = *(const v8bf*)(vp + 16);
      acc = wmma_bf16(a.v, bb.v, acc);
    }
    #pragma unroll
    for (int i = 0; i < 8; ++i){
      int mm = i + ((lane >= 16) ? 8 : 0);
      int gi = i0 + mm;
      int d = nt*16 + (lane & 15);
      if (gi < T_ && d < DV_){
        int vd = d / F_, f = d % F_;
        int ch = h*VD_ + vd;
        VoR[(((size_t)b*C_ + ch)*T_ + gi)*F_ + f] = acc[i];
      }
    }
  }
}

// ---- K8: final projection + PReLU + LN + residual -------------------------
__global__ void final_kernel(const float* __restrict__ VoR, const float* __restrict__ outp,
                             const float* __restrict__ wp, const float* __restrict__ bp,
                             const float* __restrict__ gp, const float* __restrict__ betp,
                             const float* __restrict__ aptr, float* __restrict__ dout)
{
  __shared__ float buf[C_*F_];
  __shared__ float red[256];
  int blk = blockIdx.x; int b = blk / T_, t = blk % T_;
  int tid = threadIdx.x;
  float a = aptr[0];
  const int D = C_*F_;
  for (int idx = tid; idx < D; idx += 256){
    int o = idx / F_, f = idx % F_;
    float s = bp[o];
    for (int ch = 0; ch < C_; ++ch)
      s += wp[o*C_ + ch] * VoR[(((size_t)b*C_ + ch)*T_ + t)*F_ + f];
    buf[idx] = (s >= 0.f) ? s : a*s;
  }
  __syncthreads();
  float ls = 0.f;
  for (int idx = tid; idx < D; idx += 256) ls += buf[idx];
  float mean = block_sum(ls, red) / D;
  float lv = 0.f;
  for (int idx = tid; idx < D; idx += 256){ float d = buf[idx] - mean; lv += d*d; }
  float ri = rsqrtf(block_sum(lv, red) / D + EPS_);
  for (int idx = tid; idx < D; idx += 256){
    int o = idx / F_, f = idx % F_;
    size_t gidx = (((size_t)b*C_ + o)*T_ + t)*F_ + f;
    dout[gidx] = outp[gidx] + (buf[idx] - mean)*ri*gp[o*F_ + f] + betp[o*F_ + f];
  }
}

// ---- host launcher --------------------------------------------------------
extern "C" void kernel_launch(void* const* d_in, const int* in_sizes, int n_in,
                              void* d_out, int out_size, void* d_ws, size_t ws_size,
                              hipStream_t stream)
{
  (void)in_sizes; (void)n_in; (void)out_size; (void)ws_size;
  const float* x       = (const float*)d_in[0];
  const float* h0      = (const float*)d_in[1];
  const float* c0      = (const float*)d_in[2];
  const float* g_intra = (const float*)d_in[3];
  const float* b_intra = (const float*)d_in[4];
  const float* wih_f   = (const float*)d_in[5];
  const float* whh_f   = (const float*)d_in[6];
  const float* bih_f   = (const float*)d_in[7];
  const float* bhh_f   = (const float*)d_in[8];
  const float* wih_b   = (const float*)d_in[9];
  const float* whh_b   = (const float*)d_in[10];
  const float* bih_b   = (const float*)d_in[11];
  const float* bhh_b   = (const float*)d_in[12];
  const float* wT_intra= (const float*)d_in[13];
  const float* bT_intra= (const float*)d_in[14];
  const float* g_inter = (const float*)d_in[15];
  const float* b_inter = (const float*)d_in[16];
  const float* wih_i   = (const float*)d_in[17];
  const float* whh_i   = (const float*)d_in[18];
  const float* bih_i   = (const float*)d_in[19];
  const float* bhh_i   = (const float*)d_in[20];
  const float* wT_inter= (const float*)d_in[21];
  const float* bT_inter= (const float*)d_in[22];
  const float* wq = (const float*)d_in[23]; const float* bq = (const float*)d_in[24];
  const float* gq = (const float*)d_in[25]; const float* betq = (const float*)d_in[26];
  const float* wk = (const float*)d_in[27]; const float* bk = (const float*)d_in[28];
  const float* gk = (const float*)d_in[29]; const float* betk = (const float*)d_in[30];
  const float* wv = (const float*)d_in[31]; const float* bv = (const float*)d_in[32];
  const float* gv = (const float*)d_in[33]; const float* betv = (const float*)d_in[34];
  const float* wp = (const float*)d_in[35]; const float* bp = (const float*)d_in[36];
  const float* gp = (const float*)d_in[37]; const float* betp = (const float*)d_in[38];
  const float* aptr = (const float*)d_in[39];
  float* dout = (float*)d_out;

  char* ws = (char*)d_ws;
  size_t off = 0;
  auto take = [&](size_t bytes)->char*{ char* p = ws + off; off += (bytes + 255) & ~(size_t)255; return p; };
  __bf16* UI  = (__bf16*)take((size_t)BT_*L1_*192*2);
  __bf16* WCF = (__bf16*)take((size_t)384*G_*2);
  __bf16* WCB = (__bf16*)take((size_t)384*G_*2);
  __bf16* WCI = (__bf16*)take((size_t)384*G_*2);
  float*  BCF = (float*) take((size_t)G_*4);
  float*  BCB = (float*) take((size_t)G_*4);
  float*  BCI = (float*) take((size_t)G_*4);
  float*  YI  = (float*) take((size_t)BT_*L1_*384*4);
  float*  X1  = (float*) take((size_t)B_*C_*T_*F_*4);
  __bf16* ZN2 = (__bf16*)take((size_t)BF_*C_*T_*2);
  __bf16* U2  = (__bf16*)take((size_t)BF_*L2_*192*2);
  float*  Y2  = (float*) take((size_t)BF_*L2_*H_*4);
  float*  OUTB= (float*) take((size_t)B_*C_*T_*F_*4);
  __bf16* QB  = (__bf16*)take((size_t)NHB_*T_*DQP_*2);
  __bf16* KB  = (__bf16*)take((size_t)NHB_*T_*DQP_*2);
  __bf16* VTB = (__bf16*)take((size_t)NHB_*DVP_*TP_*2);
  float*  VOR = (float*) take((size_t)B_*C_*T_*F_*4);

  {
    size_t tot = (size_t)3*384*G_ + 3*G_;
    prep_weights<<<(unsigned)((tot + 255)/256), 256, 0, stream>>>(
        wih_f, whh_f, bih_f, bhh_f, wih_b, whh_b, bih_b, bhh_b,
        wih_i, whh_i, bih_i, bhh_i, WCF, WCB, WCI, BCF, BCB, BCI);
  }
  ln_unfold_intra<<<BT_, 256, 0, stream>>>(x, g_intra, b_intra, UI);
  lstm_kernel<<<BT_/16, 256, 0, stream>>>(UI, WCF, BCF, nullptr, nullptr,
                                          YI, BT_, L1_, 384, 0,   0);
  lstm_kernel<<<BT_/16, 256, 0, stream>>>(UI, WCB, BCB, nullptr, nullptr,
                                          YI, BT_, L1_, 384, 192, 1);
  {
    size_t tot = (size_t)B_*C_*T_*F_;
    convT_res<<<(unsigned)((tot + 255)/256), 256, 0, stream>>>(
        YI, wT_intra, bT_intra, x, X1, L1_, 384, 0);
  }
  ln_inter<<<B_*F_*4, 256, 0, stream>>>(X1, g_inter, b_inter, ZN2);
  {
    size_t tot = (size_t)BF_*L2_*192;
    unfold_inter<<<(unsigned)((tot + 255)/256), 256, 0, stream>>>(ZN2, U2);
  }
  lstm_kernel<<<(BF_ + 15)/16, 256, 0, stream>>>(U2, WCI, BCI, h0, c0,
                                                 Y2, BF_, L2_, 192, 0, 0);
  {
    size_t tot = (size_t)B_*C_*T_*F_;
    convT_res<<<(unsigned)((tot + 255)/256), 256, 0, stream>>>(
        Y2, wT_inter, bT_inter, X1, OUTB, L2_, 192, 1);
  }
  proj_head_ln<<<NH_*B_*T_, 256, 0, stream>>>(OUTB, wq, bq, gq, betq, aptr, QB,  E_,  DQP_, 0, 0);
  proj_head_ln<<<NH_*B_*T_, 256, 0, stream>>>(OUTB, wk, bk, gk, betk, aptr, KB,  E_,  DQP_, 0, 0);
  proj_head_ln<<<NH_*B_*T_, 256, 0, stream>>>(OUTB, wv, bv, gv, betv, aptr, VTB, VD_, DVP_, 1, TP_);
  {
    dim3 grid((T_ + 15)/16, NHB_);
    attn_kernel<<<grid, 256, 0, stream>>>(QB, KB, VTB, VOR);
  }
  final_kernel<<<B_*T_, 256, 0, stream>>>(VOR, OUTB, wp, bp, gp, betp, aptr, dout);
}
